// Similarity_4690104287482
// MI455X (gfx1250) — matrix operand order repaired and checked by reading, not compiled, not run on hardware
//
#include <hip/hip_runtime.h>

typedef __attribute__((ext_vector_type(16))) _Float16 v16h;
typedef __attribute__((ext_vector_type(8)))  float    v8f;

#define NB   8
#define NC   16
#define NH   512
#define NW   512
#define NO   32
#define OH   256
#define OW   256
#define KTOT 288           // 16c * 3 * 3 * {x, x^2}
#define K2   (KTOT / 2)    // 144 packed u32 per pixel patch
#define PSTR 148           // per-pixel LDS pitch in u32 (padded for bank spread)

__device__ __forceinline__ unsigned short f2h(float f) {
    union { _Float16 h; unsigned short u; } cv;
    cv.h = (_Float16)f;
    return cv.u;
}
__device__ __forceinline__ unsigned int packxx(float v) {
    return (unsigned int)f2h(v) | ((unsigned int)f2h(v * v) << 16);
}

// Prologue: fold reference algebra into GEMM weights + bias.
//   k even: 2*|sw|*templ (multiplies x);  k odd: -|sw| (multiplies x^2)
//   bias[n] = -sum |sw|*templ^2
__global__ void sim_prep(const float* __restrict__ t, const float* __restrict__ s,
                         unsigned short* __restrict__ wk, float* __restrict__ bias) {
    int n = threadIdx.x;               // 32 threads, one per output channel
    float acc = 0.f;
    for (int c = 0; c < NC; ++c)
        for (int p = 0; p < 3; ++p)
            for (int q = 0; q < 3; ++q) {
                int idx = ((n * NC + c) * 3 + p) * 3 + q;
                float w  = fabsf(s[idx]);
                float tv = t[idx];
                int k2 = (c * 3 + p) * 3 + q;
                wk[n * KTOT + 2 * k2]     = f2h(2.f * w * tv);
                wk[n * KTOT + 2 * k2 + 1] = f2h(-w);
                acc += w * tv * tv;
            }
    bias[n] = -acc;
}

__global__ __launch_bounds__(128)
void sim_main(const float* __restrict__ x, const unsigned int* __restrict__ wkg,
              const float* __restrict__ bias, float* __restrict__ out) {
    __shared__ unsigned int xlds[64 * PSTR];   // im2col: [pixel][k2] packed (f16 x | f16 x^2)
    __shared__ unsigned int wlds[NO * K2];     // f16 weight pairs

    const int tid  = threadIdx.x;
    const int lane = tid & 31;
    const int wave = tid >> 5;
    const int bx   = blockIdx.x;   // pixel quarter (64 out pixels each)
    const int iy   = blockIdx.y;   // output row
    const int b    = blockIdx.z;   // batch

    const int cb = bx * 128;       // first input column of strip (SAME here: pad_lo = 0)
    const int r0 = 2 * iy;         // input rows r0 .. r0+2 (zero-pad row 512)

    // ---- stage weights to LDS ----
    for (int widx = tid; widx < NO * K2; widx += 128)
        wlds[widx] = wkg[widx];

    // ---- stage x as per-pixel im2col patches; col = tid keeps loads coalesced, no div ----
    const float* xb = x + (long)b * NC * NH * NW;
    for (int c = 0; c < NC; ++c) {
        for (int p = 0; p < 3; ++p) {
            const int gr   = r0 + p;
            const int row3 = (c * 3 + p) * 3;
            const int col  = tid;                  // 0..127 (gc <= 511 always in range)
            float v = 0.f;
            if (gr < NH)
                v = xb[((long)c * NH + gr) * NW + (cb + col)];
            unsigned int pk = packxx(v);
            if ((col & 1) == 0) {                  // even col: q=0 of pix, q=2 of pix-1
                int pix = col >> 1;
                xlds[pix * PSTR + row3 + 0] = pk;
                if (pix > 0)
                    xlds[(pix - 1) * PSTR + row3 + 2] = pk;
            } else {                               // odd col: q=1 of pix
                xlds[(col >> 1) * PSTR + row3 + 1] = pk;
            }
        }
    }
    // tail: col = 128 feeds only q=2 of pixel 63
    if (tid < NC * 3) {
        int c = tid / 3, p = tid - c * 3;
        int gr = r0 + p, gc = cb + 128;
        float v = (gr < NH && gc < NW) ? xb[((long)c * NH + gr) * NW + gc] : 0.f;
        xlds[63 * PSTR + tid * 3 + 2] = packxx(v);
    }
    __syncthreads();

    // ---- implicit GEMM: all fragment addresses affine -> ds_load_b128 pairs ----
    const int half   = lane >> 4;                 // 0: K-low half, 1: K-high half
    const int pixoff = wave * 16 + (lane & 15);   // output pixel within workgroup (0..63)
    const int m      = lane & 15;                 // A row (channel within 16-wide tile)

    const unsigned int* Bp  = &xlds[pixoff * PSTR + half * 8];
    const unsigned int* A0p = &wlds[m * K2 + half * 8];
    const unsigned int* A1p = &wlds[(m + 16) * K2 + half * 8];

    v8f c0 = {}; v8f c1 = {};
    #pragma unroll
    for (int kk = 0; kk < 9; ++kk) {              // K = 288 = 9 * 32
        union { int i[8]; v16h h; } Bf, A0, A1;
        #pragma unroll
        for (int v = 0; v < 8; ++v)
            Bf.i[v] = Bp[kk * 16 + v];            // 8 consecutive u32 -> 2x b128
        #pragma unroll
        for (int v = 0; v < 4; ++v) {             // A layout: V0-7 <- k2+0..7 (contiguous, same as B)
            A0.i[v]     = A0p[kk * 16 + v];
            A0.i[v + 4] = A0p[kk * 16 + 4 + v];
            A1.i[v]     = A1p[kk * 16 + v];
            A1.i[v + 4] = A1p[kk * 16 + 4 + v];
        }
        c0 = __builtin_amdgcn_wmma_f32_16x16x32_f16(false, A0.h, false, Bf.h, (short)0, c0, false, false);
        c1 = __builtin_amdgcn_wmma_f32_16x16x32_f16(false, A1.h, false, Bf.h, (short)0, c1, false, false);
    }

    // ---- epilogue: add bias, store (coalesced over lanes 0-15 / 16-31) ----
    const int j  = bx * 64 + pixoff;
    const int hC = half * 8;                      // C layout: lanes 16-31 hold M = v+8
    #pragma unroll
    for (int v = 0; v < 8; ++v) {
        int n0 = v + hC;
        out[((long)(b * NO + n0) * OH + iy) * OW + j]      = c0[v] + bias[n0];
        out[((long)(b * NO + n0 + 16) * OH + iy) * OW + j] = c1[v] + bias[n0 + 16];
    }
}

extern "C" void kernel_launch(void* const* d_in, const int* in_sizes, int n_in,
                              void* d_out, int out_size, void* d_ws, size_t ws_size,
                              hipStream_t stream) {
    const float* x  = (const float*)d_in[0];
    const float* t  = (const float*)d_in[1];
    const float* sw = (const float*)d_in[2];
    unsigned short* wk = (unsigned short*)d_ws;                   // 32*288 f16 = 18432 B
    float* bias = (float*)((char*)d_ws + (size_t)NO * KTOT * 2);  // 32 f32
    float* out  = (float*)d_out;

    sim_prep<<<1, 32, 0, stream>>>(t, sw, wk, bias);
    dim3 grid(4, OH, NB);   // 4 pixel-quarters x 256 rows x 8 batch = 8192 workgroups
    sim_main<<<grid, 128, 0, stream>>>(x, (const unsigned int*)wk, bias, out);
}
